// MultiScaleBKLayer_62319975465271
// MI455X (gfx1250) — matrix operand order repaired and checked by reading, not compiled, not run on hardware
//
#include <hip/hip_runtime.h>
#include <math.h>

typedef __attribute__((ext_vector_type(16))) _Float16 v16h;
typedef __attribute__((ext_vector_type(4)))  _Float16 v4h;
typedef __attribute__((ext_vector_type(8)))  float    v8f;

static constexpr int M_LIN    = 0;
static constexpr int M_GELU   = 1;
static constexpr int M_RELU   = 2;
static constexpr int M_ACCROW = 3;
static constexpr int M_BASE   = 4;

__device__ __forceinline__ float gelu_exact(float x) {
    return 0.5f * x * (1.0f + erff(x * 0.7071067811865475f));
}

template<int MODE>
__device__ __forceinline__ void epi_store(float val, int row, int col,
                                          const float* __restrict__ bias,
                                          float* __restrict__ C, int ldc,
                                          const float* __restrict__ Base,
                                          const float* __restrict__ rowScale,
                                          float sc)
{
    val += bias[col];
    const size_t o = (size_t)row * ldc + col;
    if (MODE == M_LIN)         C[o] = val;
    else if (MODE == M_GELU)   C[o] = gelu_exact(val);
    else if (MODE == M_RELU)   C[o] = fmaxf(val, 0.0f);
    else if (MODE == M_ACCROW) C[o] += rowScale[row] * val;
    else                       C[o] = Base[o] + sc * val;
}

// ---------------------------------------------------------------------------
// Tiled WMMA GEMM: C(MxN) = epilogue(A(MxK) @ B(KxN) + bias)
// Block: 256 threads = 8 wave32s arranged 4(M) x 2(N); block tile 128x64.
// Each wave owns a 2x2 fragment grid (two 16-row A frags x two 16-col B
// frags) -> 4 WMMAs per K-step from 8 ds_load_b128 (square-tile reuse).
// LDS tiles stored in *fragment order*:
//   A frag: element (r,k) -> lane = r + 16*((k>>3)&1), slot = (k&7) + 8*(k>>4)
//   B frag: element (k,n) -> lane = n + 16*((k>>3)&1), slot = (k&7) + 8*(k>>4)
// so staging writes packed v4h (ds_store_b64) and each lane reads a whole
// 16-half fragment as one 32-byte LDS read (2x ds_load_b128).
// Requires M%128==0, N%64==0, K%32==0 (true for all uses here).
// ---------------------------------------------------------------------------
template<int MODE>
__global__ __launch_bounds__(256)
void gemm_wmma_f16(const float* __restrict__ A, int lda,
                   const float* __restrict__ Bm, int ldb,
                   const float* __restrict__ bias,
                   float* __restrict__ C, int ldc,
                   const float* __restrict__ Base,
                   const float* __restrict__ rowScale,
                   const float* __restrict__ scalarScale,
                   int K)
{
    __shared__ __align__(32) _Float16 AF[8][32][16];  // 8 M-fragments (128 rows)
    __shared__ __align__(32) _Float16 BF[4][32][16];  // 4 N-fragments (64 cols)

    const int tid  = threadIdx.x;
    const int wave = tid >> 5;
    const int lane = tid & 31;
    const int wm   = wave >> 1;        // 0..3 -> rows 32*wm..+31 (A frags 2wm,2wm+1)
    const int wn   = wave & 1;         // 0..1 -> cols 32*wn..+31 (B frags 2wn,2wn+1)
    const int row0 = blockIdx.x * 128;
    const int col0 = blockIdx.y * 64;

    v8f acc[2][2] = {};

    for (int k0 = 0; k0 < K; k0 += 32) {
        // --- stage A tile 128x32 (fp32 -> f16, fragment order) ---
        // 1024 float4 tasks, 4 per thread; each writes one packed v4h.
        #pragma unroll
        for (int it = 0; it < 4; ++it) {
            const int id = tid + it * 256;       // 0..1023
            const int rr = id >> 3;              // 0..127
            const int kk = (id & 7) << 2;        // 0,4,...,28
            const float4 f = *reinterpret_cast<const float4*>(
                A + (size_t)(row0 + rr) * lda + k0 + kk);
            const int sub   = rr >> 4;
            const int laneD = (rr & 15) + (((kk >> 3) & 1) << 4);
            const int slot  = (kk & 7) + ((kk >> 4) << 3);
            v4h p; p.x = (_Float16)f.x; p.y = (_Float16)f.y;
                   p.z = (_Float16)f.z; p.w = (_Float16)f.w;
            *reinterpret_cast<v4h*>(&AF[sub][laneD][slot]) = p;
        }
        // --- stage B tile 32x64 (k-major per-thread reads, fragment order) ---
        // 512 tasks (64 n x 8 k-groups), 2 per thread; reads stay coalesced
        // across a wave (consecutive tid -> consecutive n).
        #pragma unroll
        for (int it = 0; it < 2; ++it) {
            const int id = tid + it * 256;       // 0..511
            const int n  = id & 63;
            const int kk = (id >> 6) << 2;       // 0,4,...,28
            const float* bp = Bm + (size_t)(k0 + kk) * ldb + col0 + n;
            const float f0 = bp[0];
            const float f1 = bp[(size_t)ldb];
            const float f2 = bp[(size_t)ldb * 2];
            const float f3 = bp[(size_t)ldb * 3];
            const int sub   = n >> 4;
            const int laneD = (n & 15) + (((kk >> 3) & 1) << 4);
            const int slot  = (kk & 7) + ((kk >> 4) << 3);
            v4h p; p.x = (_Float16)f0; p.y = (_Float16)f1;
                   p.z = (_Float16)f2; p.w = (_Float16)f3;
            *reinterpret_cast<v4h*>(&BF[sub][laneD][slot]) = p;
        }
        __syncthreads();

        const v16h av0 = *reinterpret_cast<const v16h*>(&AF[2 * wm][lane][0]);
        const v16h av1 = *reinterpret_cast<const v16h*>(&AF[2 * wm + 1][lane][0]);
        const v16h bv0 = *reinterpret_cast<const v16h*>(&BF[2 * wn][lane][0]);
        const v16h bv1 = *reinterpret_cast<const v16h*>(&BF[2 * wn + 1][lane][0]);
        acc[0][0] = __builtin_amdgcn_wmma_f32_16x16x32_f16(
            false, av0, false, bv0, (short)0, acc[0][0], false, false);
        acc[0][1] = __builtin_amdgcn_wmma_f32_16x16x32_f16(
            false, av0, false, bv1, (short)0, acc[0][1], false, false);
        acc[1][0] = __builtin_amdgcn_wmma_f32_16x16x32_f16(
            false, av1, false, bv0, (short)0, acc[1][0], false, false);
        acc[1][1] = __builtin_amdgcn_wmma_f32_16x16x32_f16(
            false, av1, false, bv1, (short)0, acc[1][1], false, false);
        __syncthreads();
    }

    const float sc  = (MODE == M_BASE) ? scalarScale[0] : 1.0f;
    const int  hlf  = lane >> 4;
    const int  r16  = lane & 15;
    #pragma unroll
    for (int mi = 0; mi < 2; ++mi) {
        #pragma unroll
        for (int ni = 0; ni < 2; ++ni) {
            #pragma unroll
            for (int i = 0; i < 8; ++i) {
                const int row = row0 + 32 * wm + 16 * mi + i + 8 * hlf;  // f32 C layout
                const int col = col0 + 32 * wn + 16 * ni + r16;
                epi_store<MODE>(acc[mi][ni][i], row, col,
                                bias, C, ldc, Base, rowScale, sc);
            }
        }
    }
}

// ---------------------------------------------------------------------------
// LayerNorm over row length L (+optional exact gelu). One block per row.
// ---------------------------------------------------------------------------
__global__ __launch_bounds__(256)
void layernorm_k(const float* __restrict__ in, float* __restrict__ out,
                 const float* __restrict__ g, const float* __restrict__ b,
                 int L, int doGelu)
{
    __shared__ float sh1[256], sh2[256];
    const int row = blockIdx.x;
    const int tid = threadIdx.x;
    const float* x = in + (size_t)row * L;
    float s = 0.f, s2 = 0.f;
    for (int i = tid; i < L; i += 256) { float v = x[i]; s += v; s2 += v * v; }
    sh1[tid] = s; sh2[tid] = s2; __syncthreads();
    for (int off = 128; off > 0; off >>= 1) {
        if (tid < off) { sh1[tid] += sh1[tid + off]; sh2[tid] += sh2[tid + off]; }
        __syncthreads();
    }
    const float mean = sh1[0] / (float)L;
    const float var  = sh2[0] / (float)L - mean * mean;
    const float rstd = rsqrtf(var + 1e-5f);
    float* o = out + (size_t)row * L;
    for (int i = tid; i < L; i += 256) {
        float v = (x[i] - mean) * rstd * g[i] + b[i];
        if (doGelu) v = gelu_exact(v);
        o[i] = v;
    }
}

// imp = hid(8192x512) @ dw2(512) + db2 ; one wave32 per token
__global__ __launch_bounds__(256)
void imp_stage2(const float* __restrict__ hid, const float* __restrict__ w2,
                const float* __restrict__ b2, float* __restrict__ imp, int T)
{
    int t = blockIdx.x * 8 + (threadIdx.x >> 5);
    int lane = threadIdx.x & 31;
    if (t >= T) return;
    const float* h = hid + (size_t)t * 512;
    float acc = 0.f;
    for (int c = lane; c < 512; c += 32) acc += h[c] * w2[c];
    for (int off = 16; off > 0; off >>= 1) acc += __shfl_xor(acc, off, 32);
    if (lane == 0) imp[t] = acc + b2[0];
}

// adaptive downsample pooling: xd0[b,m,d] = softmax(pool_w[m]) . x[b,2m:2m+2,d]
__global__ __launch_bounds__(256)
void pool_k(const float* __restrict__ x, const float* __restrict__ pw,
            float* __restrict__ xd0)
{
    int idx = blockIdx.x * 256 + threadIdx.x;   // over B*M*D = 4194304
    int d = idx & 1023;
    int t = idx >> 10;          // b*M + m
    int m = t & 1023;
    int b = t >> 10;
    float p0 = pw[2 * m], p1 = pw[2 * m + 1];
    float mx = fmaxf(p0, p1);
    float e0 = expf(p0 - mx), e1 = expf(p1 - mx);
    float inv = 1.0f / (e0 + e1);
    const float* xb = x + ((size_t)b * 2048 + 2 * (size_t)m) * 1024 + d;
    xd0[idx] = (e0 * inv) * xb[0] + (e1 * inv) * xb[1024];
}

// gate: probs=softmax(xd@gate_w+gate_b); top-1 -> per-expert row scale buffer
__global__ __launch_bounds__(256)
void gate_k(const float* __restrict__ xd, const float* __restrict__ gw,
            const float* __restrict__ gb, float* __restrict__ scl, int T)
{
    int t = blockIdx.x * 8 + (threadIdx.x >> 5);
    int lane = threadIdx.x & 31;
    if (t >= T) return;
    const float* xr = xd + (size_t)t * 1024;
    float a0 = 0.f, a1 = 0.f, a2 = 0.f, a3 = 0.f;
    for (int d = lane; d < 1024; d += 32) {
        float xv = xr[d];
        const float* gr = gw + (size_t)d * 4;
        a0 += xv * gr[0]; a1 += xv * gr[1]; a2 += xv * gr[2]; a3 += xv * gr[3];
    }
    for (int off = 16; off > 0; off >>= 1) {
        a0 += __shfl_xor(a0, off, 32); a1 += __shfl_xor(a1, off, 32);
        a2 += __shfl_xor(a2, off, 32); a3 += __shfl_xor(a3, off, 32);
    }
    if (lane == 0) {
        float l[4] = { a0 + gb[0], a1 + gb[1], a2 + gb[2], a3 + gb[3] };
        float mx = fmaxf(fmaxf(l[0], l[1]), fmaxf(l[2], l[3]));
        float e[4]; float s = 0.f; int best = 0;
        for (int i = 0; i < 4; ++i) { e[i] = expf(l[i] - mx); s += e[i]; }
        for (int i = 1; i < 4; ++i) if (l[i] > l[best]) best = i;
        float p = e[best] / s;
        for (int i = 0; i < 4; ++i) scl[(size_t)i * T + t] = (i == best) ? p : 0.0f;
    }
}

// v = clip(ffn @ vp_w + vp_b, +-3); score = xd @ sp_w + sp_b  (wave per token)
__global__ __launch_bounds__(256)
void vscore_k(const float* __restrict__ ffn, const float* __restrict__ xd,
              const float* __restrict__ vpw, const float* __restrict__ vpb,
              const float* __restrict__ spw, const float* __restrict__ spb,
              float* __restrict__ vout, float* __restrict__ sout, int T)
{
    int t = blockIdx.x * 8 + (threadIdx.x >> 5);
    int lane = threadIdx.x & 31;
    if (t >= T) return;
    const float* fr = ffn + (size_t)t * 1024;
    const float* xr = xd + (size_t)t * 1024;
    float av = 0.f, as = 0.f;
    for (int d = lane; d < 1024; d += 32) { av += fr[d] * vpw[d]; as += xr[d] * spw[d]; }
    for (int off = 16; off > 0; off >>= 1) {
        av += __shfl_xor(av, off, 32); as += __shfl_xor(as, off, 32);
    }
    if (lane == 0) {
        vout[t] = fminf(fmaxf(av + vpb[0], -3.0f), 3.0f);
        sout[t] = as + spb[0];
    }
}

// hard top-k mask per batch row via binary search on the threshold
__global__ __launch_bounds__(256)
void topk_k(const float* __restrict__ score, float* __restrict__ mask,
            int Mlen, int kkeep)
{
    __shared__ float rlo[256], rhi[256];
    __shared__ int cnt;
    __shared__ float slo, shi;
    int b = blockIdx.x, tid = threadIdx.x;
    const float* s = score + (size_t)b * Mlen;
    float lo = 3.0e38f, hi = -3.0e38f;
    for (int i = tid; i < Mlen; i += 256) { float v = s[i]; lo = fminf(lo, v); hi = fmaxf(hi, v); }
    rlo[tid] = lo; rhi[tid] = hi; __syncthreads();
    for (int off = 128; off > 0; off >>= 1) {
        if (tid < off) {
            rlo[tid] = fminf(rlo[tid], rlo[tid + off]);
            rhi[tid] = fmaxf(rhi[tid], rhi[tid + off]);
        }
        __syncthreads();
    }
    if (tid == 0) { slo = rlo[0] - 1.0f; shi = rhi[0] + 1.0f; }
    __syncthreads();
    for (int it = 0; it < 44; ++it) {
        if (tid == 0) cnt = 0;
        __syncthreads();
        float mid = 0.5f * (slo + shi);
        int c = 0;
        for (int i = tid; i < Mlen; i += 256) c += (s[i] >= mid) ? 1 : 0;
        atomicAdd(&cnt, c);
        __syncthreads();
        if (tid == 0) { if (cnt >= kkeep) slo = mid; else shi = mid; }
        __syncthreads();
    }
    float th = slo;
    for (int i = tid; i < Mlen; i += 256)
        mask[(size_t)b * Mlen + i] = (s[i] >= th) ? 1.0f : 0.0f;
}

// BK diagonal resolvent via forward/backward continued fractions (B=4 rows)
__global__ void bk_k(const float* __restrict__ v, const float* __restrict__ mask,
                     float* __restrict__ Carr, float* __restrict__ feats, int Mlen)
{
    int b = threadIdx.x;
    if (b >= 4) return;
    const float* vb = v + (size_t)b * Mlen;
    float* Cb = Carr + (size_t)b * Mlen * 2;
    float* Fb = feats + (size_t)b * Mlen * 2;
    const float* mb = mask + (size_t)b * Mlen;
    float cr = 0.f, ci = 0.f;
    Cb[0] = 0.f; Cb[1] = 0.f;
    for (int i = 1; i < Mlen; ++i) {          // C_i = 1/(a_{i-1} - C_{i-1})
        float ar = -vb[i - 1], ai = 1.0f;
        float dr = ar - cr, di = ai - ci;
        float den = dr * dr + di * di;
        cr = dr / den; ci = -di / den;
        Cb[2 * i] = cr; Cb[2 * i + 1] = ci;
    }
    float Dr = 0.f, Di = 0.f;                  // D_{i-1} = 1/(a_i - D_i)
    for (int i = Mlen - 1; i >= 0; --i) {
        float ar = -vb[i], ai = 1.0f;
        float gr = ar - Cb[2 * i] - Dr, gi = ai - Cb[2 * i + 1] - Di;
        float den = gr * gr + gi * gi;
        float Gr = gr / den, Gi = -gi / den;
        float mk = mb[i];
        Fb[2 * i]     = fminf(fmaxf(Gr, -10.f), 10.f) * mk;
        Fb[2 * i + 1] = fminf(fmaxf(Gi, -10.f), 10.f) * mk;
        float er = ar - Dr, ei = ai - Di;
        float dn2 = er * er + ei * ei;
        Dr = er / dn2; Di = -ei / dn2;
    }
}

// xl = ffn + bk_scale * (feats @ op_w + op_b)
__global__ __launch_bounds__(256)
void xl_k(const float* __restrict__ ffn, const float* __restrict__ feats,
          const float* __restrict__ opw, const float* __restrict__ opb,
          const float* __restrict__ bk, float* __restrict__ xl)
{
    int idx = blockIdx.x * 256 + threadIdx.x;  // B*M*D
    int d = idx & 1023;
    int t = idx >> 10;
    float fr2 = feats[2 * t], fi2 = feats[2 * t + 1];
    xl[idx] = ffn[idx] + bk[0] * (fr2 * opw[d] + fi2 * opw[1024 + d] + opb[d]);
}

// xc = x + scale_low * (upsampled t2 + pos_emb)
__global__ __launch_bounds__(256)
void xc_k(const float* __restrict__ x, const float* __restrict__ t2,
          const float* __restrict__ pos, const float* __restrict__ sl,
          float* __restrict__ xc)
{
    int idx = blockIdx.x * 256 + threadIdx.x;  // B*N*D = 8388608
    int d = idx & 1023;
    int n = (idx >> 10) & 2047;
    int b = idx >> 21;
    int m = n >> 1, j = n & 1;
    size_t t2i = ((size_t)(b * 1024 + m)) * 2048 + (size_t)j * 1024 + d;
    xc[idx] = x[idx] + sl[0] * (t2[t2i] + pos[j * 1024 + d]);
}

// ---------------------------------------------------------------------------

extern "C" void kernel_launch(void* const* d_in, const int* in_sizes, int n_in,
                              void* d_out, int out_size, void* d_ws, size_t ws_size,
                              hipStream_t stream) {
    const float* x       = (const float*)d_in[0];
    const float* dw1     = (const float*)d_in[1];
    const float* db1     = (const float*)d_in[2];
    const float* dw2     = (const float*)d_in[3];
    const float* db2     = (const float*)d_in[4];
    const float* pool_w  = (const float*)d_in[5];
    const float* dref_w  = (const float*)d_in[6];
    const float* dref_b  = (const float*)d_in[7];
    const float* dln_g   = (const float*)d_in[8];
    const float* dln_b   = (const float*)d_in[9];
    const float* gate_w  = (const float*)d_in[10];
    const float* gate_b  = (const float*)d_in[11];
    const float* ew1     = (const float*)d_in[12];
    const float* eb1     = (const float*)d_in[13];
    const float* ew2     = (const float*)d_in[14];
    const float* eb2     = (const float*)d_in[15];
    const float* vp_w    = (const float*)d_in[16];
    const float* vp_b    = (const float*)d_in[17];
    const float* sp_w    = (const float*)d_in[18];
    const float* sp_b    = (const float*)d_in[19];
    const float* op_w    = (const float*)d_in[20];
    const float* op_b    = (const float*)d_in[21];
    const float* bk_sc   = (const float*)d_in[22];
    const float* up_w1   = (const float*)d_in[23];
    const float* up_b1   = (const float*)d_in[24];
    const float* uln_g   = (const float*)d_in[25];
    const float* uln_b   = (const float*)d_in[26];
    const float* up_w2   = (const float*)d_in[27];
    const float* up_b2   = (const float*)d_in[28];
    const float* pos_emb = (const float*)d_in[29];
    const float* rln_g   = (const float*)d_in[30];
    const float* rln_b   = (const float*)d_in[31];
    const float* rw1     = (const float*)d_in[32];
    const float* rb1     = (const float*)d_in[33];
    const float* rw2     = (const float*)d_in[34];
    const float* rb2     = (const float*)d_in[35];
    const float* sc_low  = (const float*)d_in[36];
    const float* sc_ref  = (const float*)d_in[37];

    // outputs: out (B,N,D)=8388608 | imp (B,N)=8192 | mask (B,M)=4096
    float* out   = (float*)d_out;
    float* imp   = out + 8388608;
    float* maskO = out + 8388608 + 8192;

    // workspace layout (floats), with lifetime-based reuse
    float* w = (float*)d_ws;
    const size_t MF = 1024u * 1024u;
    float* hbuf  = w + 0 * MF;        // 16M : expert hidden (per e)
    float* tmp2  = w + 0 * MF;        //  8M : up GEMM1 out (after experts)
    float* tmp2b = w + 8 * MF;        //  8M : LN+gelu of tmp2
    float* hid1  = w + 16 * MF;       //  4M : importance hidden
    float* ffn   = w + 16 * MF;       //  4M : MoE output (after imp)
    float* xd0   = w + 20 * MF;       //  4M : pooled tokens
    float* xl    = w + 20 * MF;       //  4M : ffn + spec (after dref gemm)
    float* tmp1  = w + 24 * MF;       //  4M : dref GEMM out
    float* t2    = w + 24 * MF;       //  8M : up GEMM2 out (after vscore)
    float* xd    = w + 28 * MF;       //  4M : refined pooled tokens
    float* xc    = w + 32 * MF;       //  8M : combined
    float* lnxc  = w + 40 * MF;       //  8M : LN(xc)
    float* h2    = w + 48 * MF;       //  8M : refine hidden
    float* smalls = w + 56 * MF;
    float* scl   = smalls;            // 16384 : per-expert row scales
    float* vpot  = smalls + 16384;    // 4096
    float* score = smalls + 20480;    // 4096
    float* Carr  = smalls + 24576;    // 8192
    float* feats = smalls + 32768;    // 8192

    // 1) importance hidden: relu(x @ dw1 + db1)   (8192x512, K=1024)
    gemm_wmma_f16<M_RELU><<<dim3(64, 8), 256, 0, stream>>>(
        x, 1024, dw1, 512, db1, hid1, 512, nullptr, nullptr, nullptr, 1024);
    // 2) imp = hid1 @ dw2 + db2
    imp_stage2<<<1024, 256, 0, stream>>>(hid1, dw2, db2, imp, 8192);
    // 3) adaptive downsample pooling
    pool_k<<<16384, 256, 0, stream>>>(x, pool_w, xd0);
    // 4) dref GEMM (4096x1024, K=1024)
    gemm_wmma_f16<M_LIN><<<dim3(32, 16), 256, 0, stream>>>(
        xd0, 1024, dref_w, 1024, dref_b, tmp1, 1024, nullptr, nullptr, nullptr, 1024);
    // 5) LN + gelu -> xd
    layernorm_k<<<4096, 256, 0, stream>>>(tmp1, xd, dln_g, dln_b, 1024, 1);
    // 6) gate -> per-expert row scales
    gate_k<<<512, 256, 0, stream>>>(xd, gate_w, gate_b, scl, 4096);
    // 7) zero ffn accumulator
    hipMemsetAsync(ffn, 0, (size_t)4 * MF * sizeof(float), stream);
    // 8) MoE experts: h = gelu(xd@ew1[e]+eb1[e]); ffn += gate_e * (h@ew2[e]+eb2[e])
    for (int e = 0; e < 4; ++e) {
        gemm_wmma_f16<M_GELU><<<dim3(32, 64), 256, 0, stream>>>(
            xd, 1024, ew1 + (size_t)e * 1024 * 4096, 4096, eb1 + (size_t)e * 4096,
            hbuf, 4096, nullptr, nullptr, nullptr, 1024);
        gemm_wmma_f16<M_ACCROW><<<dim3(32, 16), 256, 0, stream>>>(
            hbuf, 4096, ew2 + (size_t)e * 4096 * 1024, 1024, eb2 + (size_t)e * 1024,
            ffn, 1024, nullptr, scl + (size_t)e * 4096, nullptr, 4096);
    }
    // 9) v potential + sparsity score
    vscore_k<<<512, 256, 0, stream>>>(ffn, xd, vp_w, vp_b, sp_w, sp_b, vpot, score, 4096);
    // 10) top-k mask (k = round(1024*0.4) = 410)
    topk_k<<<4, 256, 0, stream>>>(score, maskO, 1024, 410);
    // 11) BK resolvent -> clipped, masked features
    bk_k<<<1, 32, 0, stream>>>(vpot, maskO, Carr, feats, 1024);
    // 12) xl = ffn + bk_scale * (feats @ op_w + op_b)
    xl_k<<<16384, 256, 0, stream>>>(ffn, feats, op_w, op_b, bk_sc, xl);
    // 13) up GEMM1 (4096x2048, K=1024)
    gemm_wmma_f16<M_LIN><<<dim3(32, 32), 256, 0, stream>>>(
        xl, 1024, up_w1, 2048, up_b1, tmp2, 2048, nullptr, nullptr, nullptr, 1024);
    // 14) LN + gelu over 2048
    layernorm_k<<<4096, 256, 0, stream>>>(tmp2, tmp2b, uln_g, uln_b, 2048, 1);
    // 15) up GEMM2 (4096x2048, K=2048)
    gemm_wmma_f16<M_LIN><<<dim3(32, 32), 256, 0, stream>>>(
        tmp2b, 2048, up_w2, 2048, up_b2, t2, 2048, nullptr, nullptr, nullptr, 2048);
    // 16) xc = x + scale_low * (upsample(t2) + pos_emb)
    xc_k<<<32768, 256, 0, stream>>>(x, t2, pos_emb, sc_low, xc);
    // 17) LN(xc)
    layernorm_k<<<8192, 256, 0, stream>>>(xc, lnxc, rln_g, rln_b, 1024, 0);
    // 18) refine hidden: gelu(lnxc @ rw1 + rb1)  (8192x1024, K=1024)
    gemm_wmma_f16<M_GELU><<<dim3(64, 16), 256, 0, stream>>>(
        lnxc, 1024, rw1, 1024, rb1, h2, 1024, nullptr, nullptr, nullptr, 1024);
    // 19) out = xc + scale_ref * (h2 @ rw2 + rb2)
    gemm_wmma_f16<M_BASE><<<dim3(64, 16), 256, 0, stream>>>(
        h2, 1024, rw2, 1024, rb2, out, 1024, xc, nullptr, sc_ref, 1024);
}
